// Attention_77876347011327
// MI455X (gfx1250) — matrix operand order repaired and checked by reading, not compiled, not run on hardware
//
#include <hip/hip_runtime.h>
#include <hip/hip_bf16.h>

// Problem sizes (fixed by the reference)
#define BB    4
#define SS    2048
#define EE    1024      // embedding dim
#define KDIM  1024      // key dim
#define MTOT  (BB*SS)   // 8192 tokens

typedef __bf16 bf16_t;
typedef __attribute__((ext_vector_type(16))) __bf16 v16bf;
typedef __attribute__((ext_vector_type(8)))  __bf16 v8bf;
typedef __attribute__((ext_vector_type(8)))  float  v8f;
typedef __attribute__((ext_vector_type(4)))  int    v4i_t;

#define AS1 __attribute__((address_space(1)))
#define AS3 __attribute__((address_space(3)))

// ---- CDNA5 async global->LDS path (guarded; sync fallback keeps compile OK) --
#if defined(__has_builtin)
#if __has_builtin(__builtin_amdgcn_global_load_async_to_lds_b128)
#define ASYNC_LDS 1
#endif
#endif
#ifndef ASYNC_LDS
#define ASYNC_LDS 0
#endif

static __device__ __forceinline__
void stage16(const bf16_t* __restrict__ gsrc, bf16_t* __restrict__ ldst) {
#if ASYNC_LDS
    __builtin_amdgcn_global_load_async_to_lds_b128(
        (AS1 v4i_t*)const_cast<bf16_t*>(gsrc),
        (AS3 v4i_t*)ldst, 0, 0);
#else
    *(v8bf*)ldst = *(const v8bf*)gsrc;
#endif
}

static __device__ __forceinline__ void wait_async_lds() {
#if ASYNC_LDS
#if __has_builtin(__builtin_amdgcn_s_wait_asynccnt)
    __builtin_amdgcn_s_wait_asynccnt(0);
#else
    asm volatile("s_wait_asynccnt 0x0" ::: "memory");
#endif
#endif
}

// ---------------- WMMA fragment helpers (contiguous k = 16*(lane/16)+s, same for A and B) -------

// A-matrix 16x32 bf16: lane<16 -> row=lane, K = kb+0..15
//                      lane>=16 -> row=lane-16, K = kb+16..31
static __device__ __forceinline__
v16bf load_a_frag(const bf16_t* __restrict__ base, size_t lda,
                  int m0, int kb, int lane) {
    int row = m0 + (lane & 15);
    int k0  = kb + ((lane & 16) ? 16 : 0);
    const bf16_t* p = base + (size_t)row * lda + k0;
    v8bf lo = *(const v8bf*)(p);
    v8bf hi = *(const v8bf*)(p + 8);
    v16bf a;
#pragma unroll
    for (int i = 0; i < 8; ++i) { a[i] = lo[i]; a[8 + i] = hi[i]; }
    return a;
}

// B-matrix 32x16 bf16 from global: N=lane%16, K contiguous; elem(k,n)=base[n*ldb+k]
static __device__ __forceinline__
v16bf load_b_frag_g(const bf16_t* __restrict__ base, size_t ldb,
                    int n0, int kb, int lane) {
    int col = n0 + (lane & 15);
    int k0  = kb + ((lane & 16) ? 16 : 0);
    return *(const v16bf*)(base + (size_t)col * ldb + k0);
}

// B-matrix fragment from an LDS tile Bsh[64][40] (80B row stride, padded)
static __device__ __forceinline__
v16bf load_b_frag_lds(const bf16_t* __restrict__ tile, int n_local, int lane) {
    int col = n_local + (lane & 15);
    int k0  = (lane & 16) ? 16 : 0;
    const bf16_t* p = tile + col * 40 + k0;
    v8bf lo = *(const v8bf*)(p);
    v8bf hi = *(const v8bf*)(p + 8);
    v16bf b;
#pragma unroll
    for (int i = 0; i < 8; ++i) { b[i] = lo[i]; b[8 + i] = hi[i]; }
    return b;
}

static __device__ __forceinline__
v8f wmma_bf16(v16bf a, v16bf b, v8f c) {
    return __builtin_amdgcn_wmma_f32_16x16x32_bf16(
        false, a, false, b, (short)0, c, false, false);
}

// ---------------- Kernel 0: fp32 -> bf16 conversion --------------------------
__global__ void attn_cvt_bf16(const float* __restrict__ src,
                              bf16_t* __restrict__ dst, int n8) {
    int i = blockIdx.x * blockDim.x + threadIdx.x;
    if (i >= n8) return;
    const float4* s = (const float4*)src + (size_t)i * 2;
    float4 f0 = s[0], f1 = s[1];
    v8bf o;
    o[0] = (bf16_t)f0.x; o[1] = (bf16_t)f0.y; o[2] = (bf16_t)f0.z; o[3] = (bf16_t)f0.w;
    o[4] = (bf16_t)f1.x; o[5] = (bf16_t)f1.y; o[6] = (bf16_t)f1.z; o[7] = (bf16_t)f1.w;
    ((v8bf*)dst)[i] = o;
}

// ---------------- Kernel 1: projections Q/K/V = x @ W^T ----------------------
// Block = 8 waves covering 256 rows x 64 cols; weight tile (64x32) staged in
// LDS via async DMA, double-buffered; shared by all 8 waves.
// gridDim.y: 0 -> Q, 1 -> K, 2 -> V(transposed).
__global__ __launch_bounds__(256)
void attn_proj_gemm(const bf16_t* __restrict__ xb,
                    const bf16_t* __restrict__ Wqb,
                    const bf16_t* __restrict__ Wkb,
                    const bf16_t* __restrict__ Wvb,
                    bf16_t* __restrict__ Qb,
                    bf16_t* __restrict__ Kb,
                    bf16_t* __restrict__ Vtb) {
    __shared__ bf16_t Bsh[2][64][40];   // 40-elem (80B) row stride: bank-friendly
    const int lane = threadIdx.x & 31;
    const int wv   = threadIdx.x >> 5;           // 0..7
    const int nGroups = EE / 64;                 // 16
    const int m0 = (blockIdx.x / nGroups) * 256 + wv * 32;
    const int n0 = (blockIdx.x % nGroups) * 64;
    const int which = blockIdx.y;
    const bf16_t* W = (which == 0) ? Wqb : (which == 1) ? Wkb : Wvb;

    // per-thread 16B slice of the 64x32 weight tile
    const int scol = threadIdx.x >> 2;           // 0..63
    const int skp  = (threadIdx.x & 3) * 8;      // 0,8,16,24
    const bf16_t* gstage = W + (size_t)(n0 + scol) * EE + skp;

    stage16(gstage, &Bsh[0][scol][skp]);         // prefetch K-step 0

    v8f acc[2][4] = {};
    const int KT = EE / 32;                      // 32 K-steps

    // one K-step of compute out of LDS buffer (kt & 1)
    auto compute_step = [&](int kt) {
        const int kb = kt * 32;
        v16bf a0 = load_a_frag(xb, EE, m0,      kb, lane);
        v16bf a1 = load_a_frag(xb, EE, m0 + 16, kb, lane);
        const bf16_t* tile = &Bsh[kt & 1][0][0];
        v16bf bfr[4];
#pragma unroll
        for (int j = 0; j < 4; ++j)
            bfr[j] = load_b_frag_lds(tile, 16 * j, lane);
#pragma unroll
        for (int j = 0; j < 4; ++j) {
            acc[0][j] = wmma_bf16(a0, bfr[j], acc[0][j]);
            acc[1][j] = wmma_bf16(a1, bfr[j], acc[1][j]);
        }
    };

    for (int kt = 0; kt < KT - 1; ++kt) {        // steady state: branch-free body
        wait_async_lds();
        __syncthreads();                         // tile kt visible to all waves
        stage16(gstage + (kt + 1) * 32, &Bsh[(kt & 1) ^ 1][scol][skp]);
        compute_step(kt);
    }
    wait_async_lds();                            // peeled final K-step
    __syncthreads();
    compute_step(KT - 1);

    if (which < 2) {
        bf16_t* out = (which == 0) ? Qb : Kb;    // row-major [MTOT, KDIM]
#pragma unroll
        for (int i = 0; i < 2; ++i) {
#pragma unroll
            for (int j = 0; j < 4; ++j) {
                int col   = n0 + 16 * j + (lane & 15);
                int rbase = m0 + 16 * i + ((lane & 16) ? 8 : 0);
#pragma unroll
                for (int r = 0; r < 8; ++r)
                    out[(size_t)(rbase + r) * KDIM + col] = (bf16_t)acc[i][j][r];
            }
        }
    } else {
        // V transposed: Vt[f * MTOT + token] -> K-contiguous B-frags for P@V
#pragma unroll
        for (int i = 0; i < 2; ++i) {
#pragma unroll
            for (int j = 0; j < 4; ++j) {
                int col   = n0 + 16 * j + (lane & 15);
                int rbase = m0 + 16 * i + ((lane & 16) ? 8 : 0);
                v8bf pk;
#pragma unroll
                for (int r = 0; r < 8; ++r) pk[r] = (bf16_t)acc[i][j][r];
                *(v8bf*)(Vtb + (size_t)col * MTOT + rbase) = pk;
            }
        }
    }
}

// ---------------- Kernel 2: scores = (Q @ K^T) * scale, 64x64/wave -----------
__global__ __launch_bounds__(256)
void attn_scores_gemm(const bf16_t* __restrict__ Qb,
                      const bf16_t* __restrict__ Kb,
                      float* __restrict__ scores) {
    const int lane = threadIdx.x & 31;
    const int wave = (blockIdx.x * blockDim.x + threadIdx.x) >> 5;
    const int nGroups = SS / 64;              // 32
    const int m0 = (wave / nGroups) * 64;
    const int n0 = (wave % nGroups) * 64;
    if (n0 > m0 + 63) return;                 // fully above the causal diagonal
    const int b = blockIdx.y;
    const bf16_t* Q = Qb + (size_t)b * SS * KDIM;
    const bf16_t* K = Kb + (size_t)b * SS * KDIM;

    v8f acc[4][4] = {};
    for (int kb = 0; kb < KDIM; kb += 32) {
        v16bf a[4];
#pragma unroll
        for (int i = 0; i < 4; ++i)
            a[i] = load_a_frag(Q, KDIM, m0 + 16 * i, kb, lane);
#pragma unroll
        for (int j = 0; j < 4; ++j) {
            v16bf bb = load_b_frag_g(K, KDIM, n0 + 16 * j, kb, lane);
#pragma unroll
            for (int i = 0; i < 4; ++i)
                acc[i][j] = wmma_bf16(a[i], bb, acc[i][j]);
        }
    }
    const float scale = 0.03125f;             // 1/sqrt(1024)
    float* sc = scores + (size_t)b * SS * SS;
#pragma unroll
    for (int i = 0; i < 4; ++i) {
#pragma unroll
        for (int j = 0; j < 4; ++j) {
            if (n0 + 16 * j > m0 + 16 * i + 15) continue;  // tile fully masked
            int col   = n0 + 16 * j + (lane & 15);
            int rbase = m0 + 16 * i + ((lane & 16) ? 8 : 0);
#pragma unroll
            for (int r = 0; r < 8; ++r)
                sc[(size_t)(rbase + r) * SS + col] = acc[i][j][r] * scale;
        }
    }
}

// ---------------- Kernel 3: causal row softmax -> bf16 P ---------------------
__global__ void attn_softmax(const float* __restrict__ scores,
                             bf16_t* __restrict__ P) {
    const int row = blockIdx.x;               // 0 .. B*S-1
    const int q   = row & (SS - 1);
    const float* src = scores + (size_t)row * SS;
    bf16_t* dst = P + (size_t)row * SS;
    const int tid = threadIdx.x;
    const int len = q + 1;

    float vals[8];
    float mx = -3.4e38f;
#pragma unroll
    for (int it = 0; it < 8; ++it) {
        int c = tid + it * 256;
        float v = (c < len) ? src[c] : -3.4e38f;
        vals[it] = v;
        mx = fmaxf(mx, v);
    }
    __shared__ float red[256];
    red[tid] = mx; __syncthreads();
    for (int s = 128; s > 0; s >>= 1) {
        if (tid < s) red[tid] = fmaxf(red[tid], red[tid + s]);
        __syncthreads();
    }
    mx = red[0]; __syncthreads();

    float sum = 0.f;
#pragma unroll
    for (int it = 0; it < 8; ++it) {
        int c = tid + it * 256;
        float e = (c < len) ? __expf(vals[it] - mx) : 0.f;
        vals[it] = e;
        sum += e;
    }
    red[tid] = sum; __syncthreads();
    for (int s = 128; s > 0; s >>= 1) {
        if (tid < s) red[tid] += red[tid + s];
        __syncthreads();
    }
    float inv = 1.0f / red[0];
#pragma unroll
    for (int it = 0; it < 8; ++it) {
        int c = tid + it * 256;
        dst[c] = (bf16_t)(vals[it] * inv);    // zeros beyond diagonal
    }
}

// ---------------- Kernel 4: out = P @ V, 64x64/wave, causal K-truncation -----
__global__ __launch_bounds__(256)
void attn_pv_gemm(const bf16_t* __restrict__ P,
                  const bf16_t* __restrict__ Vtb,
                  float* __restrict__ out) {
    const int lane = threadIdx.x & 31;
    const int wave = (blockIdx.x * blockDim.x + threadIdx.x) >> 5;
    const int nGroups = EE / 64;              // 16
    const int m0 = (wave / nGroups) * 64;
    const int n0 = (wave % nGroups) * 64;
    const int b = blockIdx.y;
    const bf16_t* Pb = P + (size_t)b * SS * SS;
    const bf16_t* Vb = Vtb + (size_t)b * SS;  // Vt[f*MTOT + b*SS + s]

    v8f acc[4][4] = {};
    const int kmax = m0 + 64;                 // P == 0 beyond each row's q
    for (int kb = 0; kb < kmax; kb += 32) {
        v16bf a[4];
#pragma unroll
        for (int i = 0; i < 4; ++i)
            a[i] = load_a_frag(Pb, SS, m0 + 16 * i, kb, lane);
#pragma unroll
        for (int j = 0; j < 4; ++j) {
            v16bf bb = load_b_frag_g(Vb, MTOT, n0 + 16 * j, kb, lane);
#pragma unroll
            for (int i = 0; i < 4; ++i)
                acc[i][j] = wmma_bf16(a[i], bb, acc[i][j]);
        }
    }
    float* ob = out + (size_t)b * SS * EE;
#pragma unroll
    for (int i = 0; i < 4; ++i) {
#pragma unroll
        for (int j = 0; j < 4; ++j) {
            int col   = n0 + 16 * j + (lane & 15);
            int rbase = m0 + 16 * i + ((lane & 16) ? 8 : 0);
#pragma unroll
            for (int r = 0; r < 8; ++r)
                ob[(size_t)(rbase + r) * EE + col] = acc[i][j][r];
        }
    }
}

// ---------------- Launch ------------------------------------------------------
extern "C" void kernel_launch(void* const* d_in, const int* in_sizes, int n_in,
                              void* d_out, int out_size, void* d_ws, size_t ws_size,
                              hipStream_t stream) {
    const float* x  = (const float*)d_in[0];
    const float* Wq = (const float*)d_in[1];
    const float* Wk = (const float*)d_in[2];
    const float* Wv = (const float*)d_in[3];
    float* out = (float*)d_out;

    char* ws = (char*)d_ws;
    size_t off = 0;
    bf16_t* xb  = (bf16_t*)(ws + off); off += (size_t)MTOT * EE * 2;      // 16 MB
    bf16_t* Wqb = (bf16_t*)(ws + off); off += (size_t)KDIM * EE * 2;      //  2 MB
    bf16_t* Wkb = (bf16_t*)(ws + off); off += (size_t)KDIM * EE * 2;      //  2 MB
    bf16_t* Wvb = (bf16_t*)(ws + off); off += (size_t)EE   * EE * 2;      //  2 MB
    bf16_t* Qb  = (bf16_t*)(ws + off); off += (size_t)MTOT * KDIM * 2;    // 16 MB
    bf16_t* Kb  = (bf16_t*)(ws + off); off += (size_t)MTOT * KDIM * 2;    // 16 MB
    bf16_t* Vtb = (bf16_t*)(ws + off); off += (size_t)EE   * MTOT * 2;    // 16 MB
    float*  sc  = (float*) (ws + off); off += (size_t)BB * SS * SS * 4;   // 64 MB
    bf16_t* Pm  = (bf16_t*)(ws + off); off += (size_t)BB * SS * SS * 2;   // 32 MB

    // 0) fp32 -> bf16
    {
        int n8x = (MTOT * EE) / 8;
        int n8w = (KDIM * EE) / 8;
        attn_cvt_bf16<<<n8x / 256, 256, 0, stream>>>(x,  xb,  n8x);
        attn_cvt_bf16<<<n8w / 256, 256, 0, stream>>>(Wq, Wqb, n8w);
        attn_cvt_bf16<<<n8w / 256, 256, 0, stream>>>(Wk, Wkb, n8w);
        attn_cvt_bf16<<<n8w / 256, 256, 0, stream>>>(Wv, Wvb, n8w);
    }
    // 1) projections: blocks of 8 waves over (MTOT/256)x(EE/64) = 32x16, y=Q/K/V
    attn_proj_gemm<<<dim3(512, 3), 256, 0, stream>>>(xb, Wqb, Wkb, Wvb, Qb, Kb, Vtb);
    // 2) scores: per batch (SS/64)^2 = 1024 waves -> 128 blocks, y = batch
    attn_scores_gemm<<<dim3(128, BB), 256, 0, stream>>>(Qb, Kb, sc);
    // 3) softmax: one block per row
    attn_softmax<<<BB * SS, 256, 0, stream>>>(sc, Pm);
    // 4) out = P @ V: per batch (SS/64)*(EE/64) = 512 waves -> 64 blocks, y = batch
    attn_pv_gemm<<<dim3(64, BB), 256, 0, stream>>>(Pm, Vtb, out);
}
